// Attention_11845519802987
// MI455X (gfx1250) — matrix operand order repaired and checked by reading, not compiled, not run on hardware
//
#include <hip/hip_runtime.h>
#include <stdint.h>

// ---------------------------------------------------------------------------
// CDNA5 / gfx1250 attention pipeline: one fp32->bf16 conversion pass, then
// pure-bf16 WMMA GEMMs with async double-buffered global->LDS staging.
// ---------------------------------------------------------------------------

typedef __attribute__((ext_vector_type(16))) __bf16    v16bf;
typedef __attribute__((ext_vector_type(8)))  float     v8f;
typedef __attribute__((ext_vector_type(8)))  uint32_t  v8u;
typedef __attribute__((ext_vector_type(4)))  int       v4i;

typedef __attribute__((address_space(1))) v4i* gv4i_ptr;   // global int4*
typedef __attribute__((address_space(3))) v4i* lv4i_ptr;   // LDS int4*

#if defined(__has_builtin)
#  if __has_builtin(__builtin_amdgcn_global_load_async_to_lds_b128) && \
      __has_builtin(__builtin_amdgcn_s_wait_asynccnt)
#    define HAVE_ASYNC_LDS 1
#  endif
#endif
#ifndef HAVE_ASYNC_LDS
#  define HAVE_ASYNC_LDS 0
#endif

static __device__ __forceinline__ uint16_t f32_to_bf16_rne(float f) {
    uint32_t u = __builtin_bit_cast(uint32_t, f);
    uint32_t r = u + 0x7FFFu + ((u >> 16) & 1u);   // round-to-nearest-even
    return (uint16_t)(r >> 16);
}
static __device__ __forceinline__ uint32_t pack_bf16(float lo, float hi) {
    return (uint32_t)f32_to_bf16_rne(lo) | ((uint32_t)f32_to_bf16_rne(hi) << 16);
}

// 16-byte global -> LDS copy (async DMA on gfx1250, tracked by ASYNCcnt)
static __device__ __forceinline__ void copy16_g2l(const uint16_t* g, uint16_t* l) {
#if HAVE_ASYNC_LDS
    void* gnc = (void*)g;  // drop const, then retype + addrspacecast
    __builtin_amdgcn_global_load_async_to_lds_b128((gv4i_ptr)gnc, (lv4i_ptr)l, 0, 0);
#else
    *(uint4*)l = *(const uint4*)g;
#endif
}

// ---------------------------------------------------------------------------
// fp32 -> bf16 streaming conversion (n4 = element count / 4)
// ---------------------------------------------------------------------------
__global__ void __launch_bounds__(256)
cvt_f32_bf16(const float* __restrict__ in, uint16_t* __restrict__ out, int n4)
{
    int i = blockIdx.x * 256 + threadIdx.x;
    if (i < n4) {
        float4 v = ((const float4*)in)[i];
        uint2  p;
        p.x = pack_bf16(v.x, v.y);
        p.y = pack_bf16(v.z, v.w);
        ((uint2*)out)[i] = p;
    }
}

// ---------------------------------------------------------------------------
// NT GEMM:  C[M,N] = A[M,K] x B[N,K]^T (+ bias[n]);  A,B bf16 row-major over K
//   MODE 0: store bf16, transposed  CT[n*M + m]   (+bias)
//   MODE 1: store bf16, row-major   C [m*N + n]   (+bias)
//   MODE 2: store f32,  row-major   C [m*N + n]   (no bias)
// Block tile 128x128x32, 256 threads = 8 wave32, wave tile 32x64 (2x4 WMMA).
// Double-buffered LDS staging via async global->LDS DMA.
// ---------------------------------------------------------------------------
template <int MODE>
__global__ void __launch_bounds__(256)
gemm_nt_wmma(const uint16_t* __restrict__ A, const uint16_t* __restrict__ B,
             const float* __restrict__ bias, void* __restrict__ Cp,
             int M, int N, int K)
{
    constexpr int BM = 128, BN = 128, BK = 32, LDT = 40; // LDS row stride (bf16)
    __shared__ uint16_t lsA[2][BM * LDT];
    __shared__ uint16_t lsB[2][BN * LDT];

    const int tid    = threadIdx.x;
    const int lane   = tid & 31;
    const int wave   = tid >> 5;
    const int wm     = wave & 3;    // 4 waves along M, 32 rows each
    const int wn     = wave >> 2;   // 2 waves along N, 64 cols each
    const int blockM = blockIdx.y * BM;
    const int blockN = blockIdx.x * BN;
    const int lm     = lane & 15;
    const int half   = lane >> 4;

    // stage one 128x32 bf16 tile of A and B into LDS buffer `buf`
    auto stage = [&](int k0, int buf) {
#pragma unroll
        for (int it = 0; it < 2; ++it) {
            int idx = tid + it * 256;     // 512 chunks of 8 bf16 per matrix
            int row = idx >> 2;
            int c8  = (idx & 3) << 3;     // bf16 col 0,8,16,24 (16B chunks)
            copy16_g2l(&A[(size_t)(blockM + row) * K + k0 + c8],
                       &lsA[buf][row * LDT + c8]);
            copy16_g2l(&B[(size_t)(blockN + row) * K + k0 + c8],
                       &lsB[buf][row * LDT + c8]);
        }
    };

    v8f zero = {0.f, 0.f, 0.f, 0.f, 0.f, 0.f, 0.f, 0.f};
    v8f acc[2][4];
#pragma unroll
    for (int i = 0; i < 2; ++i)
#pragma unroll
        for (int j = 0; j < 4; ++j) acc[i][j] = zero;

    stage(0, 0);
    int buf = 0;
    for (int k0 = 0; k0 < K; k0 += BK) {
        const bool more = (k0 + BK) < K;
        if (more) stage(k0 + BK, buf ^ 1);     // 4 async ops into other buffer
#if HAVE_ASYNC_LDS
        if (more) __builtin_amdgcn_s_wait_asynccnt(4);  // current tile done,
        else      __builtin_amdgcn_s_wait_asynccnt(0);  // next still in flight
#endif
        __syncthreads();                       // ... for every wave

        // ---- gather WMMA fragments (ISA 7.12.2 VGPR layouts) ----
        // A 16x32: lane m=lm; V0..3 K={2v,2v+1}+8*half; V4..7 K=16+{..}+8*half
        v8u aF[2];
#pragma unroll
        for (int sm = 0; sm < 2; ++sm) {
            const uint16_t* base = &lsA[buf][(wm * 32 + sm * 16 + lm) * LDT];
#pragma unroll
            for (int v = 0; v < 8; ++v) {
                int kk = (v < 4) ? (2 * v + 8 * half) : (16 + 2 * (v - 4) + 8 * half);
                aF[sm][v] = *(const uint32_t*)&base[kk];
            }
        }
        // B 32x16: lane n=lm; V0..3 K={2v,2v+1}+8*half; V4..7 K=16+{..}+8*half
        v8u bF[4];
#pragma unroll
        for (int sn = 0; sn < 4; ++sn) {
            const uint16_t* base = &lsB[buf][(wn * 64 + sn * 16 + lm) * LDT];
#pragma unroll
            for (int v = 0; v < 8; ++v) {
                int kk = (v < 4) ? (2 * v + 8 * half) : (16 + 2 * (v - 4) + 8 * half);
                bF[sn][v] = *(const uint32_t*)&base[kk];
            }
        }

        // ---- 8x v_wmma_f32_16x16x32_bf16 ----
#pragma unroll
        for (int sm = 0; sm < 2; ++sm)
#pragma unroll
            for (int sn = 0; sn < 4; ++sn)
                acc[sm][sn] = __builtin_amdgcn_wmma_f32_16x16x32_bf16(
                    false, __builtin_bit_cast(v16bf, aF[sm]),
                    false, __builtin_bit_cast(v16bf, bF[sn]),
                    (short)0, acc[sm][sn], false, false);

        __syncthreads();   // buffer consumed; next iteration may overwrite it
        buf ^= 1;
    }

    // ---- epilogue: C layout VGPR r -> M=r+8*half, N=lm ----
#pragma unroll
    for (int sm = 0; sm < 2; ++sm) {
#pragma unroll
        for (int sn = 0; sn < 4; ++sn) {
            int gm0 = blockM + wm * 32 + sm * 16 + half * 8;
            int gn  = blockN + wn * 64 + sn * 16 + lm;
            float bv = 0.f;
            if constexpr (MODE != 2) bv = bias[gn];
#pragma unroll
            for (int r = 0; r < 8; ++r) {
                int   gm  = gm0 + r;
                float val = acc[sm][sn][r] + bv;
                if constexpr (MODE == 0)
                    ((uint16_t*)Cp)[(size_t)gn * M + gm] = f32_to_bf16_rne(val);
                else if constexpr (MODE == 1)
                    ((uint16_t*)Cp)[(size_t)gm * N + gn] = f32_to_bf16_rne(val);
                else
                    ((float*)Cp)[(size_t)gm * N + gn] = val;
            }
        }
    }
}

// ---------------------------------------------------------------------------
// Row softmax of scaled scores^T (row b over index a) -> bf16 atten^T.
// ---------------------------------------------------------------------------
__global__ void __launch_bounds__(256)
softmax_rows(const float* __restrict__ S, uint16_t* __restrict__ P,
             int cols, float scale)
{
    __shared__ float red[256];
    const int row = blockIdx.x;
    const int t   = threadIdx.x;
    const float* srow = S + (size_t)row * cols;

    float m = -3.402823466e+38f;
    for (int c = t; c < cols; c += 256) m = fmaxf(m, srow[c] * scale);
    red[t] = m; __syncthreads();
    for (int s = 128; s > 0; s >>= 1) {
        if (t < s) red[t] = fmaxf(red[t], red[t + s]);
        __syncthreads();
    }
    m = red[0]; __syncthreads();

    float sum = 0.f;
    for (int c = t; c < cols; c += 256) sum += __expf(srow[c] * scale - m);
    red[t] = sum; __syncthreads();
    for (int s = 128; s > 0; s >>= 1) {
        if (t < s) red[t] += red[t + s];
        __syncthreads();
    }
    float inv = 1.f / red[0];

    for (int c = t; c < cols; c += 256) {
        float p = __expf(srow[c] * scale - m) * inv;
        P[(size_t)row * cols + c] = f32_to_bf16_rne(p);
    }
}

// ---------------------------------------------------------------------------
extern "C" void kernel_launch(void* const* d_in, const int* in_sizes, int n_in,
                              void* d_out, int out_size, void* d_ws, size_t ws_size,
                              hipStream_t stream)
{
    const float* x  = (const float*)d_in[0];
    const float* Wq = (const float*)d_in[1];
    const float* bq = (const float*)d_in[2];
    const float* Wk = (const float*)d_in[3];
    const float* bk = (const float*)d_in[4];
    const float* Wv = (const float*)d_in[5];
    const float* bv = (const float*)d_in[6];

    constexpr int    NN   = 8192, D = 1024;
    constexpr float  NORM = 0.03125f;  // 1/sqrt(1024)
    constexpr size_t SZ_XD = (size_t)NN * D * 2;  // 16 MB (bf16 [NN,D])
    constexpr size_t SZ_DD = (size_t)D * D * 2;   //  2 MB (bf16 [D,D])

    // workspace layout (70 MB peak; scT/aT alias the dead W/x staging regions)
    char* base = (char*)d_ws;
    uint16_t* Qt  = (uint16_t*)(base + 0 * SZ_XD);          // Q^T [D,NN] bf16
    uint16_t* Kt  = (uint16_t*)(base + 1 * SZ_XD);          // K^T [D,NN] bf16
    uint16_t* Vb  = (uint16_t*)(base + 2 * SZ_XD);          // V   [NN,D] bf16
    uint16_t* xb  = (uint16_t*)(base + 3 * SZ_XD);          // x   bf16
    uint16_t* wqb = (uint16_t*)(base + 4 * SZ_XD);          // Wq  bf16
    uint16_t* wkb = (uint16_t*)(base + 4 * SZ_XD + SZ_DD);  // Wk  bf16
    uint16_t* wvb = (uint16_t*)(base + 4 * SZ_XD + 2 * SZ_DD); // Wv bf16
    float*    scT = (float*)   (base + 4 * SZ_XD);          // aliases wqb/wkb
    uint16_t* aT  = (uint16_t*)(base + 4 * SZ_XD + 2 * SZ_DD); // aliases wvb

    dim3 blk(256);
    dim3 gQKV(D / 128, NN / 128);  // 8 x 64 blocks
    dim3 gS(D / 128, D / 128);     // 8 x 8 blocks

    // one-time fp32 -> bf16 conversion (x read once instead of 3x at fp32)
    cvt_f32_bf16<<<dim3(NN * D / 4 / 256), blk, 0, stream>>>(x,  xb,  NN * D / 4);
    cvt_f32_bf16<<<dim3(D * D / 4 / 256),  blk, 0, stream>>>(Wq, wqb, D * D / 4);
    cvt_f32_bf16<<<dim3(D * D / 4 / 256),  blk, 0, stream>>>(Wk, wkb, D * D / 4);
    cvt_f32_bf16<<<dim3(D * D / 4 / 256),  blk, 0, stream>>>(Wv, wvb, D * D / 4);

    // Q^T, K^T (transposed bf16 store), V (row-major bf16)
    gemm_nt_wmma<0><<<gQKV, blk, 0, stream>>>(xb, wqb, bq, Qt, NN, D, D);
    gemm_nt_wmma<0><<<gQKV, blk, 0, stream>>>(xb, wkb, bk, Kt, NN, D, D);
    gemm_nt_wmma<1><<<gQKV, blk, 0, stream>>>(xb, wvb, bv, Vb, NN, D, D);

    // scores^T[b,a] = sum_n K^T[b,n] * Q^T[a,n]   (fp32, in dead W region)
    gemm_nt_wmma<2><<<gS, blk, 0, stream>>>(Kt, Qt, nullptr, scT, D, D, NN);

    // atten^T = row-softmax(scores^T * norm)  == reference softmax(axis=0)
    softmax_rows<<<dim3(D), blk, 0, stream>>>(scT, aT, D, NORM);

    // out[n,j] = sum_a V[n,a] * atten^T[j,a]  -> fp32 d_out
    gemm_nt_wmma<2><<<gQKV, blk, 0, stream>>>(Vb, aT, nullptr, (float*)d_out,
                                              NN, D, D);
}
